// H2DGSurvLogisticHazard_44220983280208
// MI455X (gfx1250) — matrix operand order repaired and checked by reading, not compiled, not run on hardware
//
#include <hip/hip_runtime.h>
#include <hip/hip_bf16.h>
#include <stdint.h>

// ============================================================================
// H2DGSurv forward for MI455X (gfx1250), wave32 + WMMA bf16.
//
// Key algebraic simplifications (exact, not approximations):
//  * gi(...) (identity-edge GATv2) == xs @ mean_h(Wl blocks) + mean_h(bl) + bias
//    (softmax over a single edge is 1; Wr/att/xd are dead code).
//  * head-mean commutes with the linear map -> 512x512 "effective" weights.
//  * sums of gi terms become K-concatenated single GEMMs (stacked Weff).
// Only the 3 gk relations (tma k=3, lymph k=4, tumor k=2) need real attention.
//
// All GEMMs: bf16 inputs (converted+padded on device), fp32 WMMA accumulate
// (v_wmma_f32_16x16x32_bf16), fused bias/scale/residual/ReLU/bf16 epilogue.
// Workspace requirement: ~290 MB.
// ============================================================================

typedef unsigned short u16;
typedef __bf16 v16bf __attribute__((ext_vector_type(16)));
typedef float  v8f   __attribute__((ext_vector_type(8)));

__device__ __forceinline__ u16 f2bf(float f) {           // round-to-nearest-even
  unsigned u = __float_as_uint(f);
  u += 0x7FFFu + ((u >> 16) & 1u);
  return (u16)(u >> 16);
}
__device__ __forceinline__ float bf2f(u16 v) {
  return __uint_as_float(((unsigned)v) << 16);
}

// ---------------------------------------------------------------------------
// Generic bf16 GEMM: out = epi(A[M,K] @ W[K,N]); M%64==0, N%64==0, K%32==0.
// Block: 128 threads = 4 waves in a 2x2 grid; each wave does a 32x32 tile via
// 2x2 v_wmma_f32_16x16x32_bf16. A tile staged in LDS in WMMA A-fragment order
// (lane-contiguous 32B -> 2x ds_load_b128 per fragment); B tile row-major
// (lane%16==N column, 16 k rows gathered down it per fragment).
// Epilogue: v = (acc + bias[n]) * scale; v += res[m,n]; v = relu(v) if act.
// ---------------------------------------------------------------------------
__global__ __launch_bounds__(128) void k_gemm(
    const u16* __restrict__ A, int lda,
    const u16* __restrict__ W, int ldb,
    int K,
    const float* __restrict__ bias,
    const float* __restrict__ res, int ldres,
    float scale, int act,
    float* __restrict__ outF, int ldof,
    u16* __restrict__ outB, int ldob)
{
  __shared__ __align__(32) u16 As[4][32][16];   // [16-row subtile][lane][elem]
  __shared__ __align__(32) u16 Bs[32][64];      // row-major k x n
  const int tid  = threadIdx.x;
  const int lane = tid & 31;
  const int wave = tid >> 5;
  const int wr = wave >> 1, wc = wave & 1;
  const long rowBase = (long)blockIdx.y * 64;
  const long colBase = (long)blockIdx.x * 64;

  v8f c00 = {0.f,0.f,0.f,0.f,0.f,0.f,0.f,0.f};
  v8f c01 = c00, c10 = c00, c11 = c00;

  for (int kb = 0; kb < K; kb += 32) {
    __syncthreads();
    // ---- stage A: 64 rows x 32 k as 256 (row, k-octet) units, 2 per thread.
    // octet o of row r maps to fragment (lane, elem) per ISA A-layout:
    //  o=0: k0..7  -> lane=m,   e=0..7 ; o=1: k8..15  -> lane=m+16, e=0..7
    //  o=2: k16..23-> lane=m,   e=8..15; o=3: k24..31 -> lane=m+16, e=8..15
#pragma unroll
    for (int u = 0; u < 2; ++u) {
      int unit = tid + u * 128;
      int r = unit >> 2, o = unit & 3;
      uint4 d = *reinterpret_cast<const uint4*>(A + (rowBase + r) * (long)lda + kb + o * 8);
      int s = r >> 4, m = r & 15;
      int dl = m + ((o & 1) << 4);
      int e0 = (o >= 2) ? 8 : 0;
      *reinterpret_cast<uint4*>(&As[s][dl][e0]) = d;
    }
    // ---- stage B: 32 k x 64 n as 256 (k, n-octet) units.
#pragma unroll
    for (int u = 0; u < 2; ++u) {
      int unit = tid + u * 128;
      int kk = unit >> 3, o = unit & 7;
      uint4 d = *reinterpret_cast<const uint4*>(W + (long)(kb + kk) * ldb + colBase + o * 8);
      *reinterpret_cast<uint4*>(&Bs[kk][o * 8]) = d;
    }
    __syncthreads();
    v16bf a0 = *reinterpret_cast<const v16bf*>(&As[2 * wr    ][lane][0]);
    v16bf a1 = *reinterpret_cast<const v16bf*>(&As[2 * wr + 1][lane][0]);
    v16bf b0, b1;
    {
      const int bh = lane >> 4, bn = 32 * wc + (lane & 15);
#pragma unroll
      for (int s = 0; s < 16; ++s) {
        int kq = (s < 8) ? (8 * bh + s) : (16 + 8 * bh + (s - 8));
        b0[s] = __builtin_bit_cast(__bf16, Bs[kq][bn]);
        b1[s] = __builtin_bit_cast(__bf16, Bs[kq][bn + 16]);
      }
    }
    c00 = __builtin_amdgcn_wmma_f32_16x16x32_bf16(false, a0, false, b0, (short)0, c00, false, false);
    c01 = __builtin_amdgcn_wmma_f32_16x16x32_bf16(false, a0, false, b1, (short)0, c01, false, false);
    c10 = __builtin_amdgcn_wmma_f32_16x16x32_bf16(false, a1, false, b0, (short)0, c10, false, false);
    c11 = __builtin_amdgcn_wmma_f32_16x16x32_bf16(false, a1, false, b1, (short)0, c11, false, false);
  }

  // C/D layout: VGPR r, lane<16 -> M=r, N=lane; lane>=16 -> M=r+8, N=lane-16.
  const int half = lane >> 4, nloc = lane & 15;
  const long m0 = rowBase + 32 * wr + half * 8;
  const long n0 = colBase + 32 * wc + nloc;
  float bv0 = bias ? bias[n0]      : 0.f;
  float bv1 = bias ? bias[n0 + 16] : 0.f;

  auto epi = [&](const v8f& c, long mrow, long ncol, float bv) {
#pragma unroll
    for (int r = 0; r < 8; ++r) {
      long m = mrow + r;
      float v = (c[r] + bv) * scale;
      if (res)  v += res[m * (long)ldres + ncol];
      if (act)  v = v > 0.f ? v : 0.f;
      if (outF) outF[m * (long)ldof + ncol] = v;
      if (outB) outB[m * (long)ldob + ncol] = f2bf(v);
    }
  };
  epi(c00, m0,      n0,      bv0);
  epi(c01, m0,      n0 + 16, bv1);
  epi(c10, m0 + 16, n0,      bv0);
  epi(c11, m0 + 16, n0 + 16, bv1);
}

// ---------------------------------------------------------------------------
// GATv2 attention for k leaves per patient (k<=4). One block per patient,
// wave h = head h. acc[b,d] += (1/H) * sum_h sum_j alpha_jh * xl[b*k+j, h*512+d]
// ---------------------------------------------------------------------------
__global__ __launch_bounds__(128) void k_attn(
    const u16* __restrict__ xl, const u16* __restrict__ xr,
    const float* __restrict__ att, int k, float* __restrict__ acc)
{
  __shared__ float sm[4][512];
  const int b = blockIdx.x;
  const int lane = threadIdx.x & 31;
  const int h = threadIdx.x >> 5;
  const u16* xrp = xr + (long)b * 2048 + h * 512;
  const float* ap = att + h * 512;
  float logit[4] = {0.f, 0.f, 0.f, 0.f};
  for (int j = 0; j < k; ++j) {
    const u16* xlp = xl + ((long)b * k + j) * 2048 + h * 512;
    float a = 0.f;
#pragma unroll
    for (int t = 0; t < 16; ++t) {
      int d = t * 32 + lane;
      float e = bf2f(xlp[d]) + bf2f(xrp[d]);
      e = e > 0.f ? e : 0.2f * e;             // leaky_relu(., 0.2)
      a += ap[d] * e;
    }
#pragma unroll
    for (int off = 16; off > 0; off >>= 1) a += __shfl_xor(a, off, 32);
    logit[j] = a;
  }
  float mx = -1e30f;
  for (int j = 0; j < k; ++j) mx = fmaxf(mx, logit[j]);
  float w[4]; float den = 0.f;
  for (int j = 0; j < 4; ++j) { w[j] = (j < k) ? __expf(logit[j] - mx) : 0.f; den += w[j]; }
  float inv = 1.f / den;
#pragma unroll
  for (int t = 0; t < 16; ++t) {
    int d = t * 32 + lane;
    float s = 0.f;
    for (int j = 0; j < k; ++j)
      s += w[j] * bf2f(xl[((long)b * k + j) * 2048 + h * 512 + d]);
    sm[h][d] = s * inv;
  }
  __syncthreads();
  for (int d = threadIdx.x; d < 512; d += 128)
    acc[(long)b * 512 + d] += 0.25f * (sm[0][d] + sm[1][d] + sm[2][d] + sm[3][d]);
}

// ---------------------------------------------------------------------------
// LayerNorm over D=512, fp32 in -> bf16 out (strided). One wave per row.
// ---------------------------------------------------------------------------
__global__ __launch_bounds__(256) void k_ln(
    const float* __restrict__ x, const float* __restrict__ g, const float* __restrict__ bb,
    u16* __restrict__ out, int ldo)
{
  int row  = blockIdx.x * 8 + (threadIdx.x >> 5);
  int lane = threadIdx.x & 31;
  const float* xp = x + (long)row * 512;
  float v[16]; float s = 0.f;
#pragma unroll
  for (int t = 0; t < 16; ++t) { v[t] = xp[t * 32 + lane]; s += v[t]; }
#pragma unroll
  for (int off = 16; off > 0; off >>= 1) s += __shfl_xor(s, off, 32);
  float mu = s * (1.f / 512.f);
  float var = 0.f;
#pragma unroll
  for (int t = 0; t < 16; ++t) { float d = v[t] - mu; var += d * d; }
#pragma unroll
  for (int off = 16; off > 0; off >>= 1) var += __shfl_xor(var, off, 32);
  float r = rsqrtf(var * (1.f / 512.f) + 1e-5f);
#pragma unroll
  for (int t = 0; t < 16; ++t) {
    int d = t * 32 + lane;
    out[(long)row * ldo + d] = f2bf((v[t] - mu) * r * g[d] + bb[d]);
  }
}

// ---------------- small utility kernels ----------------
// f32 -> bf16 with column padding (k >= K writes 0) and strided destination.
__global__ void k_cvt(u16* __restrict__ dst, int ldd, int dcol,
                      const float* __restrict__ src, int lsrc, int M, int K, int Kw)
{
  long i = (long)blockIdx.x * 256 + threadIdx.x;
  long tot = (long)M * Kw;
  if (i >= tot) return;
  int m = (int)(i / Kw), kk = (int)(i % Kw);
  float v = (kk < K) ? src[(long)m * lsrc + kk] : 0.f;
  dst[(long)m * ldd + dcol + kk] = f2bf(v);
}
// f32 -> bf16 with row padding (r >= R writes 0); dense C columns.
__global__ void k_cvt_rows(u16* __restrict__ dst, const float* __restrict__ src,
                           int R, int Rpad, int C)
{
  long i = (long)blockIdx.x * 256 + threadIdx.x;
  long tot = (long)Rpad * C;
  if (i >= tot) return;
  int r = (int)(i / C);
  dst[i] = (r < R) ? f2bf(src[i]) : (u16)0;
}
__global__ void k_padf(float* dst, const float* src, int n, int npad)
{
  int i = blockIdx.x * 256 + threadIdx.x;
  if (i >= npad) return;
  dst[i] = (i < n) ? src[i] : 0.f;
}
// Effective gi weight: dst[512x512] = bf16(mean over 4 head blocks of Wl[512,2048]).
__global__ void k_eff_w(u16* __restrict__ dst, const float* __restrict__ Wl)
{
  int i = blockIdx.x * 256 + threadIdx.x;
  if (i >= 512 * 512) return;
  int kk = i >> 9, n = i & 511;
  const float* p = Wl + (long)kk * 2048 + n;
  dst[i] = f2bf(0.25f * (p[0] + p[512] + p[1024] + p[1536]));
}
// Effective gi bias: mean over head blocks of bl + out-bias.
__global__ void k_eff_b(float* dst, const float* bl, const float* bias)
{
  int n = blockIdx.x * 256 + threadIdx.x;
  if (n >= 512) return;
  dst[n] = 0.25f * (bl[n] + bl[512 + n] + bl[1024 + n] + bl[1536 + n]) + bias[n];
}
__global__ void k_vsum(float* dst, const float* a, float ca, const float* b, float cb,
                       const float* c, float cc, const float* d, float cd,
                       const float* e, float ce, int n)
{
  int i = blockIdx.x * 256 + threadIdx.x;
  if (i >= n) return;
  float v = ca * a[i];
  if (b) v += cb * b[i];
  if (c) v += cc * c[i];
  if (d) v += cd * d[i];
  if (e) v += ce * e[i];
  dst[i] = v;
}
// s2 = relu(scale * (in + b0 + b1 + b2 + b3)); N fixed 512.
__global__ void k_combine_relu(float* __restrict__ out, const float* __restrict__ in,
                               const float* b0, const float* b1, const float* b2, const float* b3,
                               float scale, long total)
{
  long i = (long)blockIdx.x * 256 + threadIdx.x;
  if (i >= total) return;
  int n = (int)(i & 511);
  float v = (in[i] + b0[n] + b1[n] + b2[n] + b3[n]) * scale;
  out[i] = v > 0.f ? v : 0.f;
}
// Extract the 20 valid columns from the 64-padded final GEMM output.
__global__ void k_slice(float* __restrict__ out, const float* __restrict__ in)
{
  int i = blockIdx.x * 256 + threadIdx.x;
  if (i >= 4096 * 20) return;
  int b = i / 20, j = i % 20;
  out[i] = in[b * 64 + j];
}

// ============================================================================
extern "C" void kernel_launch(void* const* d_in, const int* in_sizes, int n_in,
                              void* d_out, int out_size, void* d_ws, size_t ws_size,
                              hipStream_t stream)
{
  (void)in_sizes; (void)n_in; (void)out_size; (void)ws_size;
  const int B = 4096;
  auto F = [&](int i) { return (const float*)d_in[i]; };

  // --- param leaf indices (JAX pytree order: sorted dict keys, lists in order)
  // inputs 0..10, then params['conv1'] (blood, clinical, history, lymph,
  // pathological, surgery_desc, surgery_report, tma, tumor; each Wl,Wr,att,bias,bl,br),
  // conv3 (self, step1..4), enc (W,b per leaf), head (W,b x3), ln (b,g per step),
  // selfc, skip, temporal.
  const int G_BLOOD=11, G_CLIN=17, G_HIST=23, G_LYMPH=29, G_PATH=35,
            G_SDESC=41, G_SREP=47, G_TMA=53, G_TUMOR=59;
  const int G3_SELF=65, G3_S1=71, G3_S2=77, G3_S3=83, G3_S4=89;
  const int E_BLOOD=95, E_CLIN=97, E_HIST=99, E_LYMPH=101, E_PATH=103,
            E_SDESC=105, E_SREP=107, E_TMA=109, E_TUMOR=111;
  const int HD_W0=113, HD_B0=114, HD_W1=115, HD_B1=116, HD_W2=117, HD_B2=118;
  const int LN1_B=119, LN1_G=120, LN2_B=121, LN2_G=122, LN3_B=123, LN3_G=124,
            LN4_B=125, LN4_G=126;
  const int G_SELFC=127, G_SKIP=133, G_TEMP=139;

  // --- workspace allocator (256B aligned slabs; total ~290 MB) ---
  char* wsBase = (char*)d_ws;
  size_t off = 0;
  auto allocB = [&](size_t elems) { u16*  p = (u16*) (wsBase + off); off = (off + elems*2 + 255) & ~size_t(255); return p; };
  auto allocF = [&](size_t elems) { float* p = (float*)(wsBase + off); off = (off + elems*4 + 255) & ~size_t(255); return p; };

  // bf16 padded inputs
  u16* xcB  = allocB((size_t)B*128);   u16* xbB  = allocB((size_t)B*96);
  u16* xpB  = allocB((size_t)B*64);    u16* xtB  = allocB((size_t)3*B*64);
  u16* xlyB = allocB((size_t)4*B*96);  u16* xtuB = allocB((size_t)2*B*96);
  u16* xhB  = allocB((size_t)B*768);   u16* xrpB = allocB((size_t)B*768);
  u16* xdsB = allocB((size_t)B*768);   u16* xs1B = allocB((size_t)B*512);
  // bf16 encoder weights (row-padded)
  u16* WencC = allocB(128*512); u16* WencB = allocB(96*512);  u16* WencP = allocB(64*512);
  u16* WencT = allocB(64*512);  u16* WencL = allocB(96*512);  u16* WencU = allocB(96*512);
  u16* WencH = allocB(768*512); u16* WencR = allocB(768*512); u16* WencD = allocB(768*512);
  // encoder outputs (bf16); hCB=[hC|hB], hRD=[hR|hD]
  u16* hCB = allocB((size_t)B*1024); u16* hP = allocB((size_t)B*512);
  u16* hT  = allocB((size_t)3*B*512); u16* hL = allocB((size_t)4*B*512);
  u16* hU  = allocB((size_t)2*B*512); u16* hH = allocB((size_t)B*512);
  u16* hRD = allocB((size_t)B*1024);
  // effective-weight stacks (bf16, [Ktot,512])
  u16* Wst_s1 = allocB(1024*512); u16* Wst_s4 = allocB(1024*512);
  u16* WeffP  = allocB(262144);   u16* WeffH  = allocB(262144);
  u16* WstL21 = allocB(262144);   u16* WstL22 = allocB(524288);
  u16* WstL23 = allocB(786432);   u16* WstL24 = allocB(1048576);
  u16* Wst_c3 = allocB((size_t)2560*512);
  // gk full weights + xl/xr (bf16, reused across the 3 relations)
  u16* WlB = allocB((size_t)512*2048); u16* WrB = allocB((size_t)512*2048);
  u16* xlbuf = allocB((size_t)4*B*2048); u16* xrbuf = allocB((size_t)B*2048);
  // layer buffers
  u16* nbuf = allocB((size_t)B*2048);  // [n1|n2|n3|n4]
  u16* cbuf = allocB((size_t)B*2560);  // [s1'|s2'|s3'|s4'|master]
  u16* hbuf = allocB((size_t)B*512);
  u16* h1 = allocB((size_t)B*256); u16* h2 = allocB((size_t)B*128);
  u16* Wh1 = allocB(512*256); u16* Wh2 = allocB(256*128); u16* Wh3 = allocB(128*64);
  // fp32 buffers
  float* s1 = allocF((size_t)B*512); float* s2 = allocF((size_t)B*512);
  float* s3 = allocF((size_t)B*512); float* s4 = allocF((size_t)B*512);
  float* Cs2 = allocF((size_t)B*512);
  float* Cout = allocF((size_t)B*64);
  float* beffC=allocF(512); float* beffB=allocF(512); float* beffP=allocF(512);
  float* beffH=allocF(512); float* beffR=allocF(512); float* beffD=allocF(512);
  float* bT=allocF(512); float* bS=allocF(512); float* bF_=allocF(512);
  float* b31=allocF(512); float* b32=allocF(512); float* b33=allocF(512);
  float* b34=allocF(512); float* b3s=allocF(512);
  float* bs1=allocF(512); float* bs4=allocF(512);
  float* bL22=allocF(512); float* bL23=allocF(512); float* bL24=allocF(512);
  float* bc3=allocF(512); float* bh3p=allocF(64);

  // --- launch helpers ---
  auto gemm = [&](const u16* A, int lda, const u16* W, int ldb, int M, int N, int K,
                  const float* bias, const float* res, int ldres, float scale, int act,
                  float* outF, int ldof, u16* outB, int ldob) {
    dim3 grid(N / 64, M / 64);
    k_gemm<<<grid, 128, 0, stream>>>(A, lda, W, ldb, K, bias, res, ldres, scale, act,
                                     outF, ldof, outB, ldob);
  };
  auto cvt = [&](u16* dst, int ldd, int dcol, const float* src, int lsrc, int M, int K, int Kw) {
    long n = (long)M * Kw;
    k_cvt<<<dim3((unsigned)((n + 255) / 256)), 256, 0, stream>>>(dst, ldd, dcol, src, lsrc, M, K, Kw);
  };
  auto cvt_rows = [&](u16* dst, const float* src, int R, int Rpad, int C) {
    long n = (long)Rpad * C;
    k_cvt_rows<<<dim3((unsigned)((n + 255) / 256)), 256, 0, stream>>>(dst, src, R, Rpad, C);
  };
  auto effw = [&](u16* dst, int gb) { k_eff_w<<<1024, 256, 0, stream>>>(dst, F(gb + 0)); };
  auto effb = [&](float* dst, int gb) { k_eff_b<<<2, 256, 0, stream>>>(dst, F(gb + 4), F(gb + 3)); };
  auto vsum = [&](float* dst, const float* a, float ca, const float* b, float cb,
                  const float* c, float cc, const float* d, float cd,
                  const float* e, float ce) {
    k_vsum<<<2, 256, 0, stream>>>(dst, a, ca, b, cb, c, cc, d, cd, e, ce, 512);
  };

  // ===================== prep: conversions & effective params =====================
  cvt(xcB, 128, 0, F(0), 104, B, 104, 128);
  cvt(xbB,  96, 0, F(1),  84, B,  84,  96);
  cvt(xpB,  64, 0, F(2),  64, B,  64,  64);
  cvt(xtB,  64, 0, F(3),  38, 3*B, 38, 64);
  cvt(xlyB, 96, 0, F(4),  72, 4*B, 72, 96);
  cvt(xtuB, 96, 0, F(5),  96, 2*B, 96, 96);
  cvt(xhB, 768, 0, F(6), 768, B, 768, 768);
  cvt(xrpB,768, 0, F(7), 768, B, 768, 768);
  cvt(xdsB,768, 0, F(8), 768, B, 768, 768);
  cvt(xs1B,512, 0, F(9) + (size_t)B*512, 512, B, 512, 512);   // x_step[1]
  cvt(cbuf,2560, 2048, F(10), 512, B, 512, 512);               // master -> cbuf cols 2048..2559

  cvt_rows(WencC, F(E_CLIN),  104, 128, 512); cvt_rows(WencB, F(E_BLOOD),  84,  96, 512);
  cvt_rows(WencP, F(E_PATH),   64,  64, 512); cvt_rows(WencT, F(E_TMA),    38,  64, 512);
  cvt_rows(WencL, F(E_LYMPH),  72,  96, 512); cvt_rows(WencU, F(E_TUMOR),  96,  96, 512);
  cvt_rows(WencH, F(E_HIST),  768, 768, 512); cvt_rows(WencR, F(E_SREP),  768, 768, 512);
  cvt_rows(WencD, F(E_SDESC), 768, 768, 512);

  const int WB512 = 512 * 512;
  effw(Wst_s1, G_CLIN);  effw(Wst_s1 + WB512, G_BLOOD);
  effw(Wst_s4, G_SREP);  effw(Wst_s4 + WB512, G_SDESC);
  effw(WeffP, G_PATH);   effw(WeffH, G_HIST);
  effw(WstL21, G_SELFC);
  effw(WstL22, G_TEMP);  effw(WstL22 + WB512, G_SELFC);
  effw(WstL23, G_SKIP);  effw(WstL23 + WB512, G_TEMP);  effw(WstL23 + 2*WB512, G_SELFC);
  effw(WstL24, G_SKIP);  effw(WstL24 + WB512, G_SKIP);
  effw(WstL24 + 2*WB512, G_TEMP); effw(WstL24 + 3*WB512, G_SELFC);
  effw(Wst_c3, G3_S1); effw(Wst_c3 + WB512, G3_S2); effw(Wst_c3 + 2*WB512, G3_S3);
  effw(Wst_c3 + 3*WB512, G3_S4); effw(Wst_c3 + 4*WB512, G3_SELF);

  effb(beffC, G_CLIN); effb(beffB, G_BLOOD); effb(beffP, G_PATH); effb(beffH, G_HIST);
  effb(beffR, G_SREP); effb(beffD, G_SDESC);
  effb(bT, G_TEMP); effb(bS, G_SKIP); effb(bF_, G_SELFC);
  effb(b31, G3_S1); effb(b32, G3_S2); effb(b33, G3_S3); effb(b34, G3_S4); effb(b3s, G3_SELF);
  vsum(bs1, beffC,1.f, beffB,1.f, nullptr,0, nullptr,0, nullptr,0);
  vsum(bs4, beffR,1.f, beffD,1.f, nullptr,0, nullptr,0, nullptr,0);
  vsum(bL22, bT,1.f, bF_,1.f, nullptr,0, nullptr,0, nullptr,0);
  vsum(bL23, bT,1.f, bS,1.f, bF_,1.f, nullptr,0, nullptr,0);
  vsum(bL24, bT,1.f, bS,2.f, bF_,1.f, nullptr,0, nullptr,0);
  vsum(bc3, b31,1.f, b32,1.f, b33,1.f, b34,1.f, b3s,1.f);

  // ===================== encoders: relu(x @ W + b) =====================
  gemm(xcB, 128, WencC, 512, B, 512, 128, F(E_CLIN+1),  nullptr,0, 1.f,1, nullptr,0, hCB,        1024);
  gemm(xbB,  96, WencB, 512, B, 512,  96, F(E_BLOOD+1), nullptr,0, 1.f,1, nullptr,0, hCB + 512,  1024);
  gemm(xpB,  64, WencP, 512, B, 512,  64, F(E_PATH+1),  nullptr,0, 1.f,1, nullptr,0, hP,          512);
  gemm(xtB,  64, WencT, 512, 3*B, 512, 64, F(E_TMA+1),  nullptr,0, 1.f,1, nullptr,0, hT,          512);
  gemm(xlyB, 96, WencL, 512, 4*B, 512, 96, F(E_LYMPH+1),nullptr,0, 1.f,1, nullptr,0, hL,          512);
  gemm(xtuB, 96, WencU, 512, 2*B, 512, 96, F(E_TUMOR+1),nullptr,0, 1.f,1, nullptr,0, hU,          512);
  gemm(xhB, 768, WencH, 512, B, 512, 768, F(E_HIST+1),  nullptr,0, 1.f,1, nullptr,0, hH,          512);
  gemm(xrpB,768, WencR, 512, B, 512, 768, F(E_SREP+1),  nullptr,0, 1.f,1, nullptr,0, hRD,        1024);
  gemm(xdsB,768, WencD, 512, B, 512, 768, F(E_SDESC+1), nullptr,0, 1.f,1, nullptr,0, hRD + 512,  1024);

  // ===================== conv1: steps =====================
  gemm(hCB, 1024, Wst_s1, 512, B, 512, 1024, bs1,   nullptr,0, 0.5f, 1, s1, 512, nullptr,0);
  gemm(hH,   512, WeffH,  512, B, 512,  512, beffH, nullptr,0, 1.f,  1, s3, 512, nullptr,0);
  gemm(hRD, 1024, Wst_s4, 512, B, 512, 1024, bs4,   nullptr,0, 0.5f, 1, s4, 512, nullptr,0);
  gemm(hP,   512, WeffP,  512, B, 512,  512, nullptr, nullptr,0, 1.f, 0, Cs2, 512, nullptr,0);

  auto do_gk = [&](const u16* hX, int kfac, int gb) {
    cvt(WlB, 2048, 0, F(gb + 0), 2048, 512, 2048, 2048);
    cvt(WrB, 2048, 0, F(gb + 1), 2048, 512, 2048, 2048);
    gemm(hX,   512, WlB, 2048, kfac * B, 2048, 512, F(gb + 4), nullptr,0, 1.f,0, nullptr,0, xlbuf, 2048);
    gemm(xs1B, 512, WrB, 2048, B,        2048, 512, F(gb + 5), nullptr,0, 1.f,0, nullptr,0, xrbuf, 2048);
    k_attn<<<B, 128, 0, stream>>>(xlbuf, xrbuf, F(gb + 2), kfac, Cs2);
  };
  do_gk(hT, 3, G_TMA);
  do_gk(hL, 4, G_LYMPH);
  do_gk(hU, 2, G_TUMOR);
  k_combine_relu<<<(B * 512 + 255) / 256, 256, 0, stream>>>(
      s2, Cs2, beffP, F(G_TMA + 3), F(G_LYMPH + 3), F(G_TUMOR + 3), 0.25f, (long)B * 512);

  // ===================== layer 2: LN + shared gi + residual =====================
  k_ln<<<512, 256, 0, stream>>>(s1, F(LN1_G), F(LN1_B), nbuf + 0,    2048);
  k_ln<<<512, 256, 0, stream>>>(s2, F(LN2_G), F(LN2_B), nbuf + 512,  2048);
  k_ln<<<512, 256, 0, stream>>>(s3, F(LN3_G), F(LN3_B), nbuf + 1024, 2048);
  k_ln<<<512, 256, 0, stream>>>(s4, F(LN4_G), F(LN4_B), nbuf + 1536, 2048);
  // s_i' = s_i + (nbuf[:, :Ki] @ Wstack + bsum) / r  -> cbuf (bf16)
  gemm(nbuf, 2048, WstL21, 512, B, 512,  512, bF_,  s1, 512, 1.f,       0, nullptr,0, cbuf + 0,    2560);
  gemm(nbuf, 2048, WstL22, 512, B, 512, 1024, bL22, s2, 512, 0.5f,      0, nullptr,0, cbuf + 512,  2560);
  gemm(nbuf, 2048, WstL23, 512, B, 512, 1536, bL23, s3, 512, 1.f/3.f,   0, nullptr,0, cbuf + 1024, 2560);
  gemm(nbuf, 2048, WstL24, 512, B, 512, 2048, bL24, s4, 512, 0.25f,     0, nullptr,0, cbuf + 1536, 2560);

  // ===================== conv3 + head =====================
  gemm(cbuf, 2560, Wst_c3, 512, B, 512, 2560, bc3, nullptr,0, 0.2f, 1, nullptr,0, hbuf, 512);

  cvt(Wh1, 256, 0, F(HD_W0), 256, 512, 256, 256);
  gemm(hbuf, 512, Wh1, 256, B, 256, 512, F(HD_B0), nullptr,0, 1.f, 1, nullptr,0, h1, 256);
  cvt(Wh2, 128, 0, F(HD_W1), 128, 256, 128, 128);
  gemm(h1, 256, Wh2, 128, B, 128, 256, F(HD_B1), nullptr,0, 1.f, 1, nullptr,0, h2, 128);
  cvt(Wh3, 64, 0, F(HD_W2), 20, 128, 20, 64);                 // pad N 20 -> 64
  k_padf<<<1, 256, 0, stream>>>(bh3p, F(HD_B2), 20, 64);
  gemm(h2, 128, Wh3, 64, B, 64, 128, bh3p, nullptr,0, 1.f, 0, Cout, 64, nullptr,0);

  k_slice<<<(4096 * 20 + 255) / 256, 256, 0, stream>>>((float*)d_out, Cout);
}
